// GlobalContextBranch_6992206758431
// MI455X (gfx1250) — compile-verified, matrix operand order checked by reading, not run on hardware
//
#include <hip/hip_runtime.h>
#include <hip/hip_bf16.h>
#include <math.h>

// ---------------------------------------------------------------------------
// MI455X (gfx1250) sparse-conv + SE-gate + fusion block.
//
// Roofline: feats (128MB) + conv accumulator (128MB) fit in the 192MB L2 ->
// HBM traffic ~1 pass (~400MB => ~17us floor @ 23.3TB/s); binding resources
// are L2/atomic bandwidth and matrix throughput.  gfx1250 has no wide f32
// WMMA (only 16x16x4), so both GEMMs use V_WMMA_F32_16X16X32_BF16 with a
// bf16 hi/lo split (hi*hi + hi*lo + lo*hi, f32 accumulate): 6 WMMAs per
// 16x16x64 tile at bf16 rate, ~fp32 accuracy (error ~2^-16 relative).
//
// Weights (16KB tiles) are DMA'd into LDS with the Tensor Data Mover
// (tensor_load_to_lds + s_wait_tensorcnt), then each lane pre-builds its
// B fragments in VGPRs once -> the tile loop has zero LDS traffic.
// ---------------------------------------------------------------------------

#define N_PTS 500000
#define KVOL  27
#define CH    64
#define EPS_  1e-5f

typedef __attribute__((ext_vector_type(16))) __bf16 v16bf;
typedef __attribute__((ext_vector_type(8)))  float  v8f;
typedef __attribute__((ext_vector_type(4)))  unsigned tdm_u4;
typedef __attribute__((ext_vector_type(8)))  int      tdm_i8;
typedef __attribute__((ext_vector_type(4)))  int      tdm_i4;

static __device__ __forceinline__ v8f wmma_bf16(v16bf a, v16bf b, v8f c) {
  // D = A(16x32 bf16) * B(32x16 bf16) + C(16x16 f32)
  return __builtin_amdgcn_wmma_f32_16x16x32_bf16(
      /*neg_a=*/false, a, /*neg_b=*/false, b,
      /*c_mod=*/(short)0, c, /*reuse_a=*/false, /*reuse_b=*/false);
}

// TDM: DMA a 64x64 f32 tile (16KB) from global memory into LDS.
// D# per CDNA5 ISA: group0 = {count=1, lds_addr,
// global_addr[56:0], type=2}; group1 = {data_size=4B, tensor 64x64,
// tile 64x64, dim0_stride=64}.  Tracked with TENSORcnt.
// This toolchain's builtin takes 6 args (g0,g1,g2,g3,g4,cpol); the unused
// higher-dim descriptor groups are zero-filled (2D tensor).
static __device__ __forceinline__ void tdm_load_w64x64(const float* gsrc,
                                                       float* lds_dst) {
  unsigned lds_off = (unsigned)(size_t)(void*)lds_dst;  // generic low 32 = LDS offset
  unsigned long long ga = (unsigned long long)(size_t)gsrc;
  tdm_u4 g0;
  g0[0] = 1u;                                        // count=1 (user descriptor)
  g0[1] = lds_off;                                   // lds_addr
  g0[2] = (unsigned)(ga & 0xFFFFFFFFu);              // global_addr[31:0]
  g0[3] = (unsigned)((ga >> 32) & 0x01FFFFFFu)       // global_addr[56:32]
        | (2u << 30);                                // type=2 ("image")
  tdm_i8 g1;
  g1[0] = 0x20000;                                   // data_size=2 (4 bytes)
  g1[1] = 64 << 16;                                  // tensor_dim0 = 64
  g1[2] = 64 << 16;                                  // tensor_dim1 = 64
  g1[3] = 64 << 16;                                  // tile_dim0   = 64
  g1[4] = 64;                                        // tile_dim1   = 64
  g1[5] = 64;                                        // tensor_dim0_stride = 64
  g1[6] = 0;
  g1[7] = 0;
  tdm_i4 gz4 = {0, 0, 0, 0};
  tdm_i8 gz8 = {0, 0, 0, 0, 0, 0, 0, 0};
  __builtin_amdgcn_tensor_load_to_lds(g0, g1, gz4, gz4, gz8, 0);
}

// Build the 8 (nt,kb) B fragments (hi and lo) for this lane from the raw
// f32 weight tile in LDS.  B layout mirrors the 16-bit A layout: lane<16
// holds K%32 in {0-7,16-23}, lane>=16 in {8-15,24-31}; col = nt*16 + lane%16.
static __device__ __forceinline__ void build_b_frags(const float* wstage,
                                                     int cbase, int mloc,
                                                     v16bf bh[8], v16bf bl[8]) {
#pragma unroll
  for (int f = 0; f < 8; ++f) {
    const int nt = f >> 1, kb = f & 1;
#pragma unroll
    for (int e = 0; e < 16; ++e) {
      int kk = kb * 32 + cbase + (e < 8 ? e : e + 8);
      float w = wstage[kk * CH + nt * 16 + mloc];
      __bf16 h = (__bf16)w;
      bh[f][e] = h;
      bl[f][e] = (__bf16)(w - (float)h);
    }
  }
}

// ------------------------------- kernels -----------------------------------

__global__ void zero_f32(float* __restrict__ p, size_t n) {
  size_t i = (size_t)blockIdx.x * blockDim.x + threadIdx.x;
  size_t stride = (size_t)gridDim.x * blockDim.x;
  for (; i < n; i += stride) p[i] = 0.0f;
}

// Sparse conv: for offset k, gather 16 rows of feats, GEMM against w1[k],
// scatter-add into accum with f32 atomics.
__global__ __launch_bounds__(128) void sparse_conv_wmma(
    const float* __restrict__ feats,
    const int*   __restrict__ in_idx,
    const int*   __restrict__ out_idx,
    const float* __restrict__ w1,
    float*       __restrict__ accum,
    int mmax, int tilesPerK)
{
  __shared__ float wstage[CH * CH];

  const int k = blockIdx.x;
  if (threadIdx.x == 0)                       // TDM ignores EXEC; 1 issue/block
    tdm_load_w64x64(w1 + (size_t)k * CH * CH, wstage);
  __builtin_amdgcn_s_wait_tensorcnt(0);
  __syncthreads();

  const int lane  = threadIdx.x & 31;
  const int half  = lane >> 4;       // 0: K%32 in {0-7,16-23}; 1: {8-15,24-31}
  const int mloc  = lane & 15;       // row (A) / col (B,C,D) within tile
  const int cbase = half * 8;
  const int wave  = threadIdx.x >> 5;
  const int nwv   = blockDim.x >> 5;
  const int tstep = gridDim.y * nwv;

  // B fragments are loop-invariant: build once into registers (128 VGPRs).
  v16bf bh[8], bl[8];
  build_b_frags(wstage, cbase, mloc, bh, bl);

  const int* iix = in_idx  + (size_t)k * mmax;
  const int* oix = out_idx + (size_t)k * mmax;

  for (int t = blockIdx.y * nwv + wave; t < tilesPerK; t += tstep) {
    // padded tail: sentinel entries are contiguous at the end of each k-list
    if (oix[t * 16] >= N_PTS) continue;
    if (t + tstep < tilesPerK) __builtin_prefetch(iix + (t + tstep) * 16, 0, 3);

    // ---- gather A fragments (16x64 f32 -> bf16 hi/lo, 2 K-blocks of 32) ---
    int grow = t * 16 + mloc;
    int ri = (grow < mmax) ? iix[grow] : N_PTS;
    bool valid = ri < N_PTS;                    // sentinel row contributes 0
    const float* src = feats + (size_t)ri * CH;

    v16bf ahi[2], alo[2];
#pragma unroll
    for (int kb = 0; kb < 2; ++kb) {
      float vals[16];
      if (valid) {
        float4 p0 = *(const float4*)(src + kb * 32 + cbase);
        float4 p1 = *(const float4*)(src + kb * 32 + cbase + 4);
        float4 p2 = *(const float4*)(src + kb * 32 + cbase + 16);
        float4 p3 = *(const float4*)(src + kb * 32 + cbase + 20);
        vals[0]=p0.x; vals[1]=p0.y; vals[2]=p0.z;  vals[3]=p0.w;
        vals[4]=p1.x; vals[5]=p1.y; vals[6]=p1.z;  vals[7]=p1.w;
        vals[8]=p2.x; vals[9]=p2.y; vals[10]=p2.z; vals[11]=p2.w;
        vals[12]=p3.x; vals[13]=p3.y; vals[14]=p3.z; vals[15]=p3.w;
      } else {
#pragma unroll
        for (int e = 0; e < 16; ++e) vals[e] = 0.0f;
      }
#pragma unroll
      for (int e = 0; e < 16; ++e) {
        __bf16 h = (__bf16)vals[e];
        ahi[kb][e] = h;
        alo[kb][e] = (__bf16)(vals[e] - (float)h);
      }
    }

    // ---- 4 N-tiles of 16 cols, 6 WMMAs each (hi/lo split) ----------------
    v8f acc[4];
#pragma unroll
    for (int nt = 0; nt < 4; ++nt) {
      v8f a = {};
      a = wmma_bf16(ahi[0], bh[nt * 2 + 0], a);
      a = wmma_bf16(ahi[1], bh[nt * 2 + 1], a);
      a = wmma_bf16(ahi[0], bl[nt * 2 + 0], a);
      a = wmma_bf16(ahi[1], bl[nt * 2 + 1], a);
      a = wmma_bf16(alo[0], bh[nt * 2 + 0], a);
      a = wmma_bf16(alo[1], bh[nt * 2 + 1], a);
      acc[nt] = a;
    }

    // ---- scatter-add: lane holds rows M = half*8 + r, col = nt*16 + mloc --
#pragma unroll
    for (int r = 0; r < 8; ++r) {
      int mrow = t * 16 + half * 8 + r;
      int oi = (mrow < mmax) ? oix[mrow] : N_PTS;
      if (oi < N_PTS) {
        float* dst = accum + (size_t)oi * CH + mloc;
        atomicAdd(dst +  0, acc[0][r]);
        atomicAdd(dst + 16, acc[1][r]);
        atomicAdd(dst + 32, acc[2][r]);
        atomicAdd(dst + 48, acc[3][r]);
      }
    }
  }
}

// BN1 + ReLU in-place on accum, plus per-batch pooled sums/counts
// (LDS ds_add_f32 partials, then 1 global atomic per channel per block).
__global__ __launch_bounds__(256) void bn_relu_pool(
    float* __restrict__ accum, const int* __restrict__ batch_idx,
    const float* __restrict__ g, const float* __restrict__ b,
    const float* __restrict__ m, const float* __restrict__ v,
    float* __restrict__ ctx_sum, float* __restrict__ ctx_cnt, int nrows)
{
  __shared__ float ls[4 * CH];
  __shared__ float lc[4];
  const int tid = threadIdx.x;
  if (tid < 4 * CH) ls[tid] = 0.0f;
  if (tid < 4)      lc[tid] = 0.0f;
  __syncthreads();

  const int c  = tid & (CH - 1);
  const int rs = tid >> 6;                 // 0..3
  const int row0 = blockIdx.x * 64;
  const int rend = (row0 + 64 < nrows) ? (row0 + 64) : nrows;
  const float sc = g[c] * rsqrtf(v[c] + EPS_);
  const float mm = m[c], bb = b[c];

  for (int r = row0 + rs; r < rend; r += 4) {
    int bi = batch_idx[r];
    float x = accum[(size_t)r * CH + c];
    float y = fmaxf((x - mm) * sc + bb, 0.0f);
    accum[(size_t)r * CH + c] = y;
    atomicAdd(&ls[bi * CH + c], y);
    if (c == 0) atomicAdd(&lc[bi], 1.0f);
  }
  __syncthreads();
  if (tid < 4 * CH) { float s = ls[tid]; if (s != 0.0f) atomicAdd(&ctx_sum[tid], s); }
  if (tid < 4)      { float s = lc[tid]; if (s != 0.0f) atomicAdd(&ctx_cnt[tid], s); }
}

// Tiny SE gate MLP: ctx(4x64) -> relu(ctx@ga+ga_b)(4x16) -> sigmoid(...@gb+gb_b)
__global__ __launch_bounds__(64) void gate_mlp(
    const float* __restrict__ ctx_sum, const float* __restrict__ ctx_cnt,
    const float* __restrict__ ga_w, const float* __restrict__ ga_b,
    const float* __restrict__ gb_w, const float* __restrict__ gb_b,
    float* __restrict__ gate)
{
  __shared__ float sctx[4][CH];
  __shared__ float sh[4][16];
  const int tid = threadIdx.x;                 // 64 threads
  for (int b = 0; b < 4; ++b) sctx[b][tid] = ctx_sum[b * CH + tid] / ctx_cnt[b];
  __syncthreads();
  if (tid < 16) {
    for (int b = 0; b < 4; ++b) {
      float h = ga_b[tid];
      for (int c = 0; c < CH; ++c) h += sctx[b][c] * ga_w[c * 16 + tid];
      sh[b][tid] = fmaxf(h, 0.0f);
    }
  }
  __syncthreads();
  for (int b = 0; b < 4; ++b) {
    float gq = gb_b[tid];
    for (int j = 0; j < 16; ++j) gq += sh[b][j] * gb_w[j * CH + tid];
    gate[b * CH + tid] = 1.0f / (1.0f + expf(-gq));
  }
}

// Fused: out = BN2( (feat * gate[batch]) @ wf ), WMMA bf16 hi/lo GEMM.
__global__ __launch_bounds__(128) void gate_gemm_bn2_wmma(
    const float* __restrict__ feat, const int* __restrict__ batch_idx,
    const float* __restrict__ gate, const float* __restrict__ wf,
    const float* __restrict__ g2, const float* __restrict__ b2,
    const float* __restrict__ m2, const float* __restrict__ v2,
    float* __restrict__ out, int ntiles)
{
  __shared__ float wstage[CH * CH];
  if (threadIdx.x == 0) tdm_load_w64x64(wf, wstage);
  __builtin_amdgcn_s_wait_tensorcnt(0);
  __syncthreads();

  const int lane  = threadIdx.x & 31;
  const int half  = lane >> 4;
  const int mloc  = lane & 15;
  const int cbase = half * 8;
  const int wave  = threadIdx.x >> 5;
  const int nwv   = blockDim.x >> 5;

  v16bf bh[8], bl[8];
  build_b_frags(wstage, cbase, mloc, bh, bl);

  for (int t = blockIdx.x * nwv + wave; t < ntiles; t += gridDim.x * nwv) {
    int row = t * 16 + mloc;                    // N divisible by 16
    int bi = batch_idx[row];
    const float* src = feat + (size_t)row * CH;
    const float* gp  = gate + bi * CH;

    v16bf ahi[2], alo[2];
#pragma unroll
    for (int kb = 0; kb < 2; ++kb) {
      float vals[16];
      float4 p0 = *(const float4*)(src + kb * 32 + cbase);
      float4 p1 = *(const float4*)(src + kb * 32 + cbase + 4);
      float4 p2 = *(const float4*)(src + kb * 32 + cbase + 16);
      float4 p3 = *(const float4*)(src + kb * 32 + cbase + 20);
      float4 q0 = *(const float4*)(gp  + kb * 32 + cbase);
      float4 q1 = *(const float4*)(gp  + kb * 32 + cbase + 4);
      float4 q2 = *(const float4*)(gp  + kb * 32 + cbase + 16);
      float4 q3 = *(const float4*)(gp  + kb * 32 + cbase + 20);
      vals[0]=p0.x*q0.x;  vals[1]=p0.y*q0.y;  vals[2]=p0.z*q0.z;   vals[3]=p0.w*q0.w;
      vals[4]=p1.x*q1.x;  vals[5]=p1.y*q1.y;  vals[6]=p1.z*q1.z;   vals[7]=p1.w*q1.w;
      vals[8]=p2.x*q2.x;  vals[9]=p2.y*q2.y;  vals[10]=p2.z*q2.z;  vals[11]=p2.w*q2.w;
      vals[12]=p3.x*q3.x; vals[13]=p3.y*q3.y; vals[14]=p3.z*q3.z;  vals[15]=p3.w*q3.w;
#pragma unroll
      for (int e = 0; e < 16; ++e) {
        __bf16 h = (__bf16)vals[e];
        ahi[kb][e] = h;
        alo[kb][e] = (__bf16)(vals[e] - (float)h);
      }
    }

    v8f acc[4];
#pragma unroll
    for (int nt = 0; nt < 4; ++nt) {
      v8f a = {};
      a = wmma_bf16(ahi[0], bh[nt * 2 + 0], a);
      a = wmma_bf16(ahi[1], bh[nt * 2 + 1], a);
      a = wmma_bf16(ahi[0], bl[nt * 2 + 0], a);
      a = wmma_bf16(ahi[1], bl[nt * 2 + 1], a);
      a = wmma_bf16(alo[0], bh[nt * 2 + 0], a);
      a = wmma_bf16(alo[1], bh[nt * 2 + 1], a);
      acc[nt] = a;
    }

#pragma unroll
    for (int nt = 0; nt < 4; ++nt) {
      const int col = nt * 16 + mloc;
      const float sc = g2[col] * rsqrtf(v2[col] + EPS_);
      const float mm = m2[col], bb = b2[col];
#pragma unroll
      for (int r = 0; r < 8; ++r) {
        int orow = t * 16 + half * 8 + r;
        out[(size_t)orow * CH + col] = (acc[nt][r] - mm) * sc + bb;
      }
    }
  }
}

// ------------------------------ launcher -----------------------------------

extern "C" void kernel_launch(void* const* d_in, const int* in_sizes, int n_in,
                              void* d_out, int out_size, void* d_ws, size_t ws_size,
                              hipStream_t stream) {
  const float* feats     = (const float*)d_in[0];
  const int*   batch_idx = (const int*)  d_in[1];
  const int*   in_idx    = (const int*)  d_in[2];
  const int*   out_idx   = (const int*)  d_in[3];
  const float* w1        = (const float*)d_in[4];
  const float* bn1_g     = (const float*)d_in[5];
  const float* bn1_b     = (const float*)d_in[6];
  const float* bn1_m     = (const float*)d_in[7];
  const float* bn1_v     = (const float*)d_in[8];
  const float* ga_w      = (const float*)d_in[9];
  const float* ga_b      = (const float*)d_in[10];
  const float* gb_w      = (const float*)d_in[11];
  const float* gb_b      = (const float*)d_in[12];
  const float* wf        = (const float*)d_in[13];
  const float* bn2_g     = (const float*)d_in[14];
  const float* bn2_b     = (const float*)d_in[15];
  const float* bn2_m     = (const float*)d_in[16];
  const float* bn2_v     = (const float*)d_in[17];
  float* out = (float*)d_out;

  const int mmax      = in_sizes[2] / KVOL;
  const int tilesPerK = (mmax + 15) / 16;

  // workspace layout (whole accum lives in L2 on MI455X)
  float* accum   = (float*)d_ws;                       // (N+1) x 64
  float* ctx_sum = accum + (size_t)(N_PTS + 1) * CH;   // 4 x 64
  float* ctx_cnt = ctx_sum + 4 * CH;                   // 4
  float* gatep   = ctx_cnt + 4;                        // 4 x 64

  size_t zcount = (size_t)(N_PTS + 1) * CH + 4 * CH + 4;
  zero_f32<<<2048, 256, 0, stream>>>(accum, zcount);

  sparse_conv_wmma<<<dim3(KVOL, 512), 128, 0, stream>>>(
      feats, in_idx, out_idx, w1, accum, mmax, tilesPerK);

  bn_relu_pool<<<(N_PTS + 63) / 64, 256, 0, stream>>>(
      accum, batch_idx, bn1_g, bn1_b, bn1_m, bn1_v, ctx_sum, ctx_cnt, N_PTS);

  gate_mlp<<<1, 64, 0, stream>>>(ctx_sum, ctx_cnt, ga_w, ga_b, gb_w, gb_b, gatep);

  gate_gemm_bn2_wmma<<<1024, 128, 0, stream>>>(
      accum, batch_idx, gatep, wf, bn2_g, bn2_b, bn2_m, bn2_v, out, N_PTS / 16);
}
